// E2EMaskOptWrapper_42640435314988
// MI455X (gfx1250) — compile-verified, matrix operand order checked by reading, not run on hardware
//
#include <hip/hip_runtime.h>
#include <hip/hip_bf16.h>
#include <math.h>

// ---------------- problem constants (from reference) ----------------
#define NN 50000
#define FF 512
#define PP 128
#define GG 256
#define HH 256
#define TT 24
#define EE 800000

typedef __attribute__((ext_vector_type(16))) __bf16 v16bf;
typedef __attribute__((ext_vector_type(8)))  float  v8f;

// =====================================================================
// Weight repack: f32 row-major [K x C]  ->  bf16 WMMA B-fragments.
// Fragment (jt, kt): 32 lanes x 16 bf16. lane&15 = column within 16-col
// tile; lane>>4 selects K half (0..15 vs 16..31) (contiguous k order, matching the A loads).
// out[((jt*ktiles + kt)*32 + lane)*16 + i] = W[kt*32 + (lane>>4)*16 + i][jt*16 + (lane&15)]
// =====================================================================
__global__ void k_pack_w(const float* __restrict__ W, __bf16* __restrict__ out,
                         int K, int C) {
    int id = blockIdx.x * 256 + threadIdx.x;
    if (id >= K * C) return;
    int i    = id & 15;
    int lane = (id >> 4) & 31;
    int rest = id >> 9;
    int ktiles = K >> 5;
    int kt = rest % ktiles;
    int jt = rest / ktiles;
    int row = kt * 32 + ((lane >> 4) << 4) + i;
    int col = jt * 16 + (lane & 15);
    out[id] = (__bf16)W[(size_t)row * C + col];
}

// =====================================================================
// GEMM1: p = relu((x * feat_gate) @ W_proj + b_proj), out bf16 [N x 128]
// block = 256 thr = 8 waves; wave w -> rows [blk*128 + w*16, +16)
// =====================================================================
__global__ void __launch_bounds__(256)
k_gemm1(const float* __restrict__ x, const float* __restrict__ gate,
        const __bf16* __restrict__ Wp, const float* __restrict__ bias,
        __bf16* __restrict__ p) {
    const int wave = threadIdx.x >> 5;
    const int lane = threadIdx.x & 31;
    const int rowbase = blockIdx.x * 128 + wave * 16;
    int rowA = rowbase + (lane & 15);
    if (rowA >= NN) rowA = NN - 1;              // clamp: EXEC must stay full for WMMA
    const int khalf = (lane >> 4) << 4;

    v8f acc[8];
#pragma unroll
    for (int j = 0; j < 8; ++j)
#pragma unroll
        for (int v = 0; v < 8; ++v) acc[j][v] = 0.0f;

#pragma unroll 4
    for (int kt = 0; kt < 16; ++kt) {           // K = 512 -> 16 chunks of 32
        const int kb = kt * 32 + khalf;
        const float4* xr = (const float4*)(x + (size_t)rowA * FF + kb);
        const float4* gr = (const float4*)(gate + kb);
        float vals[16];
#pragma unroll
        for (int q = 0; q < 4; ++q) {
            float4 xv = xr[q], gv = gr[q];
            vals[q * 4 + 0] = xv.x * gv.x;
            vals[q * 4 + 1] = xv.y * gv.y;
            vals[q * 4 + 2] = xv.z * gv.z;
            vals[q * 4 + 3] = xv.w * gv.w;
        }
        v16bf a;
#pragma unroll
        for (int i = 0; i < 16; ++i) a[i] = (__bf16)vals[i];

#pragma unroll
        for (int jt = 0; jt < 8; ++jt) {
            const v16bf b = *(const v16bf*)(Wp + ((size_t)((jt * 16 + kt) * 32 + lane) << 4));
            acc[jt] = __builtin_amdgcn_wmma_f32_16x16x32_bf16(
                false, a, false, b, (short)0, acc[jt], false, false);
        }
    }
    // C/D layout: VGPR v, lanes 0-15 -> M=v; lanes 16-31 -> M=v+8; N = lane&15
#pragma unroll
    for (int jt = 0; jt < 8; ++jt) {
        const int col = jt * 16 + (lane & 15);
        const float bv = bias[col];
#pragma unroll
        for (int v = 0; v < 8; ++v) {
            const int r = rowbase + v + ((lane >> 4) << 3);
            if (r < NN) {
                float val = acc[jt][v] + bv;
                p[(size_t)r * PP + col] = (__bf16)(val > 0.0f ? val : 0.0f);
            }
        }
    }
}

// =====================================================================
// Generic bf16 GEMM: out_f32[N x 256] = A_bf16[N x K] @ Wp.
// grid.y selects 128-col half (jtbase = blockIdx.y*8). No bias/act
// (GCN bias+relu applied after aggregation).
// =====================================================================
__global__ void __launch_bounds__(256)
k_gemm_b(const __bf16* __restrict__ A, const __bf16* __restrict__ Wp,
         float* __restrict__ out, int K) {
    const int wave = threadIdx.x >> 5;
    const int lane = threadIdx.x & 31;
    const int rowbase = blockIdx.x * 128 + wave * 16;
    int rowA = rowbase + (lane & 15);
    if (rowA >= NN) rowA = NN - 1;
    const int khalf = (lane >> 4) << 4;
    const int ktiles = K >> 5;
    const int jtbase = blockIdx.y * 8;

    v8f acc[8];
#pragma unroll
    for (int j = 0; j < 8; ++j)
#pragma unroll
        for (int v = 0; v < 8; ++v) acc[j][v] = 0.0f;

    for (int kt = 0; kt < ktiles; ++kt) {
        const v16bf a = *(const v16bf*)(A + (size_t)rowA * K + kt * 32 + khalf);
#pragma unroll
        for (int jt = 0; jt < 8; ++jt) {
            const v16bf b = *(const v16bf*)(Wp +
                ((size_t)(((jtbase + jt) * ktiles + kt) * 32 + lane) << 4));
            acc[jt] = __builtin_amdgcn_wmma_f32_16x16x32_bf16(
                false, a, false, b, (short)0, acc[jt], false, false);
        }
    }
#pragma unroll
    for (int jt = 0; jt < 8; ++jt) {
        const int col = (jtbase + jt) * 16 + (lane & 15);
#pragma unroll
        for (int v = 0; v < 8; ++v) {
            const int r = rowbase + v + ((lane >> 4) << 3);
            if (r < NN) out[(size_t)r * GG + col] = acc[jt][v];
        }
    }
}

// ---------------- degree / normalization ----------------
__global__ void k_deg_init(float* deg) {
    int n = blockIdx.x * 256 + threadIdx.x;
    if (n < NN) deg[n] = 1.0f;                       // self-loop weight
}
__global__ void k_deg_accum(const int* __restrict__ ei,
                            const float* __restrict__ eg, float* deg) {
    int e = blockIdx.x * 256 + threadIdx.x;
    if (e < EE) atomicAdd(&deg[ei[EE + e]], eg[e]);  // dst
}
__global__ void k_deg_fin(float* deg) {
    int n = blockIdx.x * 256 + threadIdx.x;
    if (n < NN) deg[n] = rsqrtf(deg[n]);             // store dinv in place
}

// ---------------- conv1 aggregation ----------------
// self-loop term initializes the accumulator (full overwrite -> no memset)
__global__ void k_self(const float* __restrict__ hA, const float* __restrict__ dinv,
                       float* __restrict__ agg) {
    int id = blockIdx.x * 256 + threadIdx.x;         // N*256 threads
    int n = id >> 8;
    float di = dinv[n];
    agg[id] = hA[id] * di * di;
}
// one thread = (edge, 8-feature chunk)
__global__ void k_scatter(const int* __restrict__ ei, const float* __restrict__ eg,
                          const float* __restrict__ dinv,
                          const float* __restrict__ hA, float* __restrict__ agg) {
    int id = blockIdx.x * 256 + threadIdx.x;         // E*32 threads
    int e = id >> 5;
    int c = (id & 31) << 3;
    int s = ei[e], d = ei[EE + e];
    float norm = dinv[s] * eg[e] * dinv[d];
    const float* hs = hA + (size_t)s * GG + c;
    float* od = agg + (size_t)d * GG + c;
#pragma unroll
    for (int k = 0; k < 8; ++k) atomicAdd(od + k, norm * hs[k]);
}
__global__ void k_relu_bias_bf16(const float* __restrict__ agg,
                                 const float* __restrict__ b,
                                 __bf16* __restrict__ h1) {
    int id = blockIdx.x * 256 + threadIdx.x;         // N*256 threads
    float v = agg[id] + b[id & 255];
    h1[id] = (__bf16)(v > 0.0f ? v : 0.0f);
}

// ---------------- conv2: only the target row is ever consumed ----------------
__global__ void k_tgt_init(const float* __restrict__ hB, const float* __restrict__ dinv,
                           const int* __restrict__ tgt, float* __restrict__ h2t) {
    int f = threadIdx.x;
    int t = tgt[0];
    float di = dinv[t];
    h2t[f] = hB[(size_t)t * GG + f] * di * di;
}
__global__ void k_tgt_scatter(const int* __restrict__ ei, const float* __restrict__ eg,
                              const float* __restrict__ dinv,
                              const float* __restrict__ hB,
                              const int* __restrict__ tgt, float* __restrict__ h2t) {
    int e = blockIdx.x * 256 + threadIdx.x;
    if (e >= EE) return;
    int t = tgt[0];
    if (ei[EE + e] != t) return;
    int s = ei[e];
    float norm = dinv[s] * eg[e] * dinv[t];
    const float* hs = hB + (size_t)s * GG;
    for (int f = 0; f < GG; ++f) atomicAdd(&h2t[f], norm * hs[f]);
}

// ---------------- LSTM + attention + MLP head (single block) ----------------
__global__ void __launch_bounds__(256)
k_head(const float* __restrict__ cached, const __bf16* __restrict__ h1,
       const float* __restrict__ h2t, const float* __restrict__ b_g2,
       const float* __restrict__ W_ih, const float* __restrict__ W_hh,
       const float* __restrict__ b_ih, const float* __restrict__ b_hh,
       const float* __restrict__ W_att, const float* __restrict__ b_att,
       const float* __restrict__ W_p1, const float* __restrict__ b_p1,
       const float* __restrict__ W_p2, const float* __restrict__ b_p2,
       const int* __restrict__ tgt, const int* __restrict__ epos_p,
       float* __restrict__ out) {
    __shared__ float s_tgt[2 * GG];          // target embedding / later ctx
    __shared__ float s_x[2 * GG];            // current LSTM input
    __shared__ float s_rnn[TT][HH];
    __shared__ float s_h[HH], s_c[HH];
    __shared__ float s_att[TT];
    __shared__ float s_hid[64];

    const int j = threadIdx.x;
    const int t0 = tgt[0];
    const int epos = epos_p[0];

    s_tgt[j] = (float)h1[(size_t)t0 * GG + j];
    float v2 = h2t[j] + b_g2[j];
    s_tgt[GG + j] = v2 > 0.0f ? v2 : 0.0f;
    s_h[j] = 0.0f;
    s_c[j] = 0.0f;
    __syncthreads();

    for (int t = 0; t < TT; ++t) {
        if (t == epos) { s_x[j] = s_tgt[j]; s_x[GG + j] = s_tgt[GG + j]; }
        else           { s_x[j] = cached[t * 2 * GG + j];
                         s_x[GG + j] = cached[t * 2 * GG + GG + j]; }
        __syncthreads();
        float g[4];
#pragma unroll
        for (int q = 0; q < 4; ++q) {
            const int col = q * HH + j;
            float acc = b_ih[col] + b_hh[col];
            for (int k = 0; k < 2 * GG; ++k) acc += s_x[k] * W_ih[(size_t)k * 4 * HH + col];
            for (int k = 0; k < HH; ++k)     acc += s_h[k] * W_hh[(size_t)k * 4 * HH + col];
            g[q] = acc;
        }
        __syncthreads();                      // all done reading s_h / s_x
        const float ig = 1.0f / (1.0f + __expf(-g[0]));
        const float fg = 1.0f / (1.0f + __expf(-g[1]));
        const float og = 1.0f / (1.0f + __expf(-g[3]));
        const float cc = fg * s_c[j] + ig * tanhf(g[2]);
        const float hh = og * tanhf(cc);
        s_c[j] = cc; s_h[j] = hh; s_rnn[t][j] = hh;
        __syncthreads();
    }
    if (j < TT) {
        float a = b_att[0];
        for (int k = 0; k < HH; ++k) a += s_rnn[j][k] * W_att[k];
        s_att[j] = tanhf(a);
    }
    __syncthreads();
    if (j == 0) {
        float m = -1e30f;
        for (int t = 0; t < TT; ++t) m = s_att[t] > m ? s_att[t] : m;
        float s = 0.0f;
        for (int t = 0; t < TT; ++t) { float e = __expf(s_att[t] - m); s_att[t] = e; s += e; }
        float inv = 1.0f / s;
        for (int t = 0; t < TT; ++t) s_att[t] *= inv;
    }
    __syncthreads();
    float ctx = 0.0f;
    for (int t = 0; t < TT; ++t) ctx += s_rnn[t][j] * s_att[t];
    s_tgt[j] = ctx;
    __syncthreads();
    if (j < 64) {
        float a = b_p1[j];
        for (int k = 0; k < HH; ++k) a += s_tgt[k] * W_p1[k * 64 + j];
        s_hid[j] = a > 0.0f ? a : 0.0f;
    }
    __syncthreads();
    if (j == 0) {
        float r = b_p2[0];
        for (int k = 0; k < 64; ++k) r += s_hid[k] * W_p2[k];
        out[0] = (r > 20.0f) ? r : log1pf(__expf(r));
    }
}

// =====================================================================
extern "C" void kernel_launch(void* const* d_in, const int* in_sizes, int n_in,
                              void* d_out, int out_size, void* d_ws, size_t ws_size,
                              hipStream_t stream) {
    const float* x     = (const float*)d_in[0];
    const int*   ei    = (const int*)  d_in[1];
    const float* fgate = (const float*)d_in[2];
    const float* egate = (const float*)d_in[3];
    const float* cach  = (const float*)d_in[4];
    const float* Wproj = (const float*)d_in[5];
    const float* bproj = (const float*)d_in[6];
    const float* Wg1   = (const float*)d_in[7];
    const float* bg1   = (const float*)d_in[8];
    const float* Wg2   = (const float*)d_in[9];
    const float* bg2   = (const float*)d_in[10];
    const float* Wih   = (const float*)d_in[11];
    const float* Whh   = (const float*)d_in[12];
    const float* bih   = (const float*)d_in[13];
    const float* bhh   = (const float*)d_in[14];
    const float* Watt  = (const float*)d_in[15];
    const float* batt  = (const float*)d_in[16];
    const float* Wp1   = (const float*)d_in[17];
    const float* bp1   = (const float*)d_in[18];
    const float* Wp2   = (const float*)d_in[19];
    const float* bp2   = (const float*)d_in[20];
    const int*   tgt   = (const int*)  d_in[21];
    const int*   epos  = (const int*)  d_in[22];
    float* out = (float*)d_out;

    // ---- workspace carve-up ----
    char* base = (char*)d_ws;
    size_t off = 0;
    auto carve = [&](size_t bytes) { char* p = base + off; off = (off + bytes + 255) & ~(size_t)255; return p; };
    __bf16* wpP  = (__bf16*)carve((size_t)FF * PP * 2);       // packed W_proj
    __bf16* wp1  = (__bf16*)carve((size_t)PP * GG * 2);       // packed W_g1
    __bf16* wp2  = (__bf16*)carve((size_t)GG * GG * 2);       // packed W_g2
    __bf16* p    = (__bf16*)carve((size_t)NN * PP * 2);       // projected feats (bf16)
    float*  bufA = (float*) carve((size_t)NN * GG * 4);       // hA, later reused as hB
    float*  agg  = (float*) carve((size_t)NN * GG * 4);       // conv1 accumulator
    __bf16* h1   = (__bf16*)carve((size_t)NN * GG * 2);       // relu(conv1) (bf16)
    float*  dinv = (float*) carve((size_t)NN * 4);
    float*  h2t  = (float*) carve((size_t)GG * 4);

    const dim3 B(256);
    const int mt = (NN + 127) / 128;                          // 391 row tiles

    // 1) repack weights to WMMA B-fragment layout
    k_pack_w<<<(FF * PP + 255) / 256, B, 0, stream>>>(Wproj, wpP, FF, PP);
    k_pack_w<<<(PP * GG + 255) / 256, B, 0, stream>>>(Wg1,   wp1, PP, GG);
    k_pack_w<<<(GG * GG + 255) / 256, B, 0, stream>>>(Wg2,   wp2, GG, GG);

    // 2) symmetric-norm degrees
    k_deg_init <<<(NN + 255) / 256, B, 0, stream>>>(dinv);
    k_deg_accum<<<(EE + 255) / 256, B, 0, stream>>>(ei, egate, dinv);
    k_deg_fin  <<<(NN + 255) / 256, B, 0, stream>>>(dinv);

    // 3) p = relu((x*fg) @ W_proj + b)           [WMMA]
    k_gemm1<<<mt, B, 0, stream>>>(x, fgate, wpP, bproj, p);

    // 4) hA = p @ W_g1                           [WMMA]
    k_gemm_b<<<dim3(mt, 2), B, 0, stream>>>(p, wp1, bufA, PP);

    // 5) conv1 aggregate: self + edge scatter, then bias+relu -> h1 (bf16)
    k_self        <<<(NN * GG) / 256, B, 0, stream>>>(bufA, dinv, agg);
    k_scatter     <<<(EE * 32) / 256, B, 0, stream>>>(ei, egate, dinv, bufA, agg);
    k_relu_bias_bf16<<<(NN * GG) / 256, B, 0, stream>>>(agg, bg1, h1);

    // 6) hB = h1 @ W_g2 (reuse bufA)             [WMMA]
    k_gemm_b<<<dim3(mt, 2), B, 0, stream>>>(h1, wp2, bufA, GG);

    // 7) conv2 aggregation — target row only
    k_tgt_init   <<<1, B, 0, stream>>>(bufA, dinv, tgt, h2t);
    k_tgt_scatter<<<(EE + 255) / 256, B, 0, stream>>>(ei, egate, dinv, bufA, tgt, h2t);

    // 8) LSTM + attention + MLP + softplus
    k_head<<<1, B, 0, stream>>>(cach, h1, h2t, bg2, Wih, Whh, bih, bhh,
                                Watt, batt, Wp1, bp1, Wp2, bp2, tgt, epos, out);
}
